// GradientLayer_87076166959576
// MI455X (gfx1250) — matrix operand order repaired and checked by reading, not compiled, not run on hardware
//
#include <hip/hip_runtime.h>
#include <math.h>

typedef __bf16 bf16;
typedef __attribute__((ext_vector_type(16))) __bf16 v16bf;
typedef __attribute__((ext_vector_type(8)))  __bf16 v8bf;
typedef __attribute__((ext_vector_type(8)))  float  v8f;

#define H_DIM   256
#define NBF_L   (H_DIM * H_DIM)       // bf16 elems per packed hidden layer
#define NBF_O   (8 * 512)             // packed Wo panel (256 x 16, cols>=2 zero)

// ---------------------------------------------------------------------------
// Prep: convert W1..W3 (f32, row-major K x N) and Wo (256x2, padded to 16 cols)
// into bf16, pre-tiled into the WMMA B-fragment lane layout:
//   tile (kt,nt) is 32(K) x 16(N); lane L holds column N = nt*16 + (L&15),
//   rows 16*(L>>4) + {2v, 2v+1} of the k-tile in halves of VGPR v (v=0..7).
// Each lane's 16 bf16 (32 bytes) stored contiguously -> one v16bf per lane.
// ---------------------------------------------------------------------------
__global__ void prep_weights(const float* __restrict__ W1,
                             const float* __restrict__ W2,
                             const float* __restrict__ W3,
                             const float* __restrict__ Wo,
                             bf16* __restrict__ Wq) {
  int e = blockIdx.x * blockDim.x + threadIdx.x;
  if (e >= 3 * NBF_L + NBF_O) return;
  if (e < 3 * NBF_L) {
    int l      = e / NBF_L;
    int r      = e % NBF_L;
    int tile   = r >> 9;                 // 512 bf16 per 32x16 tile
    int within = r & 511;
    int kt   = tile >> 4, nt = tile & 15;
    int lane = within >> 4, q = within & 15;
    int v    = q >> 1, hh = q & 1;
    int row  = 32 * kt + 16 * (lane >> 4) + 2 * v + hh;   // K index
    int col  = 16 * nt + (lane & 15);                     // N index
    const float* W = (l == 0) ? W1 : (l == 1) ? W2 : W3;
    Wq[e] = (bf16)W[row * H_DIM + col];
  } else {
    int r      = e - 3 * NBF_L;
    int kt     = r >> 9;
    int within = r & 511;
    int lane = within >> 4, q = within & 15;
    int v    = q >> 1, hh = q & 1;
    int row  = 32 * kt + 16 * (lane >> 4) + 2 * v + hh;
    int col  = lane & 15;
    Wq[e] = (col < 2) ? (bf16)Wo[row * 2 + col] : (bf16)0.f;
  }
}

// ---------------------------------------------------------------------------
// Fused forward + Jacobian + diagonal-Hessian kernel.
// Block = 256 threads = 8 waves; handles 16 samples.
// State vectors (rows of 16 samples x 256 units, bf16 in LDS):
//   v=0: value, v=1..4: tangents t0..t3, v=5..7: 2nd-derivs s0..s2.
// ---------------------------------------------------------------------------
__launch_bounds__(256, 1)
__global__ void mlp_hess(const float* __restrict__ x,
                         const float* __restrict__ W0, const float* __restrict__ b0,
                         const bf16*  __restrict__ Wq,
                         const float* __restrict__ b1, const float* __restrict__ b2,
                         const float* __restrict__ b3,
                         const float* __restrict__ bo,
                         float* __restrict__ out, int B) {
  __shared__ __align__(32) bf16 state[8 * 16 * H_DIM];   // 64 KB

  const int tid  = threadIdx.x;
  const int lane = tid & 31;
  const int wv   = tid >> 5;            // wave id == owned state-vector id
  const int s0   = blockIdx.x * 16;
  const float bsel = (wv == 0) ? 1.f : 0.f;   // bias feeds the value rows only

  // ---- layer 0: z = x@W0 + b0 (K=4, VALU), tangents = rows of W0,
  //      2nd derivs = 0; apply tanh chain, write bf16 state vectorized.
  {
    const int s  = tid >> 4;             // sample
    const int u0 = (tid & 15) * 16;      // 16 contiguous units per thread
    const float xr0 = x[(s0 + s) * 4 + 0], xr1 = x[(s0 + s) * 4 + 1];
    const float xr2 = x[(s0 + s) * 4 + 2], xr3 = x[(s0 + s) * 4 + 3];
#pragma unroll
    for (int half = 0; half < 2; ++half) {
      const int ub = u0 + half * 8;
      v8bf hv, tv[4], sv[3];
#pragma unroll
      for (int e = 0; e < 8; ++e) {
        int u = ub + e;
        float w0u = W0[u], w1u = W0[H_DIM + u];
        float w2u = W0[2 * H_DIM + u], w3u = W0[3 * H_DIM + u];
        float z = xr0 * w0u + xr1 * w1u + xr2 * w2u + xr3 * w3u + b0[u];
        float h = tanhf(z);
        float g = 1.f - h * h;
        hv[e] = (bf16)h;
        float zt[4] = {w0u, w1u, w2u, w3u};
#pragma unroll
        for (int j = 0; j < 4; ++j) tv[j][e] = (bf16)(g * zt[j]);
#pragma unroll
        for (int j = 0; j < 3; ++j)
          sv[j][e] = (bf16)(-2.f * h * g * zt[j] * zt[j]);
      }
      *(v8bf*)&state[(0 * 16 + s) * H_DIM + ub] = hv;
#pragma unroll
      for (int j = 0; j < 4; ++j)
        *(v8bf*)&state[((1 + j) * 16 + s) * H_DIM + ub] = tv[j];
#pragma unroll
      for (int j = 0; j < 3; ++j)
        *(v8bf*)&state[((5 + j) * 16 + s) * H_DIM + ub] = sv[j];
    }
  }
  __syncthreads();

  // ---- hidden layers 1..3: WMMA GEMM (M=16 samples, K=N=256) per state vec.
  for (int l = 0; l < 3; ++l) {
    const bf16*  Wl = Wq + l * NBF_L;
    const float* bl = (l == 0) ? b1 : (l == 1) ? b2 : b3;
    __builtin_prefetch(Wq + (l + 1) * NBF_L, 0, 1);   // next layer / Wo panel

    // Snapshot this wave's A fragments (its own 16 rows, all 8 k-tiles).
    // A 16x32 in contiguous k order (as B is packed): lanes 0-15 hold K off 0..15,
    // lanes 16-31 hold K off 16..31 (contiguous 16B LDS loads).
    v16bf a[8];
    {
      const int off  = (lane < 16) ? 0 : 16;
      const int mrow = (wv * 16 + (lane & 15)) * H_DIM;
#pragma unroll
      for (int kt = 0; kt < 8; ++kt) {
        v8bf lo = *(const v8bf*)&state[mrow + kt * 32 + off];
        v8bf hi = *(const v8bf*)&state[mrow + kt * 32 + off + 8];
        a[kt] = __builtin_shufflevector(lo, hi, 0, 1, 2, 3, 4, 5, 6, 7,
                                        8, 9, 10, 11, 12, 13, 14, 15);
      }
    }
    __syncthreads();

#pragma unroll 4
    for (int nt = 0; nt < 16; ++nt) {
      // Bias for this tile, loaded in-loop (L2-resident; scheduler hoists it;
      // no indexed-register array -> no v_movrels/M0 traffic).
      float binit = bsel * bl[nt * 16 + (lane & 15)];

      // Preload all 8 B fragments (clause + overlapped waits), then WMMAs
      // back-to-back on the accumulator chain.
      v16bf bfr[8];
#pragma unroll
      for (int kt = 0; kt < 8; ++kt)
        bfr[kt] = *(const v16bf*)(Wl + (kt * 16 + nt) * 512 + lane * 16);

      v8f acc;
#pragma unroll
      for (int r = 0; r < 8; ++r) acc[r] = binit;
#pragma unroll
      for (int kt = 0; kt < 8; ++kt)
        acc = __builtin_amdgcn_wmma_f32_16x16x32_bf16(
            false, a[kt], false, bfr[kt], (short)0, acc, false, false);

      // write pre-activation z back (in place; this wave's rows only).
      // C layout: VGPR r, lane L -> row r + 8*(L>=16), col L&15.
      const int col   = nt * 16 + (lane & 15);
      const int rbase = (lane < 16) ? 0 : 8;
#pragma unroll
      for (int r = 0; r < 8; ++r)
        state[(wv * 16 + rbase + r) * H_DIM + col] = (bf16)acc[r];
    }
    __syncthreads();   // all 8 state vectors' z complete

    // Elementwise tanh-chain across state vectors, vectorized b128 LDS I/O.
    // Thread owns (sample s, 16 contiguous units): read-all then write-all.
    {
      const int s  = tid >> 4;
      const int u0 = (tid & 15) * 16;
#pragma unroll
      for (int half = 0; half < 2; ++half) {
        const int ub = u0 + half * 8;
        v8bf zvv = *(const v8bf*)&state[(0 * 16 + s) * H_DIM + ub];
        v8bf ztv[4], zsv[3];
#pragma unroll
        for (int j = 0; j < 4; ++j)
          ztv[j] = *(const v8bf*)&state[((1 + j) * 16 + s) * H_DIM + ub];
#pragma unroll
        for (int j = 0; j < 3; ++j)
          zsv[j] = *(const v8bf*)&state[((5 + j) * 16 + s) * H_DIM + ub];
        v8bf hv, tv[4], sv[3];
#pragma unroll
        for (int e = 0; e < 8; ++e) {
          float h = tanhf((float)zvv[e]);
          float g = 1.f - h * h;
          hv[e] = (bf16)h;
#pragma unroll
          for (int j = 0; j < 4; ++j) tv[j][e] = (bf16)(g * (float)ztv[j][e]);
#pragma unroll
          for (int j = 0; j < 3; ++j) {
            float t = (float)ztv[j][e];
            sv[j][e] = (bf16)(g * (float)zsv[j][e] - 2.f * h * g * t * t);
          }
        }
        *(v8bf*)&state[(0 * 16 + s) * H_DIM + ub] = hv;
#pragma unroll
        for (int j = 0; j < 4; ++j)
          *(v8bf*)&state[((1 + j) * 16 + s) * H_DIM + ub] = tv[j];
#pragma unroll
        for (int j = 0; j < 3; ++j)
          *(v8bf*)&state[((5 + j) * 16 + s) * H_DIM + ub] = sv[j];
      }
    }
    __syncthreads();
  }

  // ---- output layer via WMMA against zero-padded Wo panel (256 x 16).
  float* resf = reinterpret_cast<float*>(state);
  {
    const bf16* Wol = Wq + 3 * NBF_L;
    v16bf a[8];
    {
      const int off  = (lane < 16) ? 0 : 16;
      const int mrow = (wv * 16 + (lane & 15)) * H_DIM;
#pragma unroll
      for (int kt = 0; kt < 8; ++kt) {
        v8bf lo = *(const v8bf*)&state[mrow + kt * 32 + off];
        v8bf hi = *(const v8bf*)&state[mrow + kt * 32 + off + 8];
        a[kt] = __builtin_shufflevector(lo, hi, 0, 1, 2, 3, 4, 5, 6, 7,
                                        8, 9, 10, 11, 12, 13, 14, 15);
      }
    }
    v16bf bfr[8];
#pragma unroll
    for (int kt = 0; kt < 8; ++kt)
      bfr[kt] = *(const v16bf*)(Wol + kt * 512 + lane * 16);

    v8f acc;
#pragma unroll
    for (int r = 0; r < 8; ++r) acc[r] = 0.f;
#pragma unroll
    for (int kt = 0; kt < 8; ++kt)
      acc = __builtin_amdgcn_wmma_f32_16x16x32_bf16(
          false, a[kt], false, bfr[kt], (short)0, acc, false, false);

    __syncthreads();     // everyone's A-frag reads done; safe to reuse state
    // C tile: row = sample, cols 0/1 = dots vs Wo columns. slot = v*2 + o.
    const int col   = lane & 15;
    const int rbase = (lane < 16) ? 0 : 8;
    if (col < 2) {
#pragma unroll
      for (int r = 0; r < 8; ++r)
        resf[(rbase + r) * 16 + wv * 2 + col] = acc[r];
    }
  }
  __syncthreads();

  if (tid < 16) {
    int s = tid, gs = s0 + s;
    const float* rs = &resf[s * 16];
    float c   = rs[0] + bo[0];
    float Fi  = rs[1] + bo[1];
    float cj0 = rs[2], cj1 = rs[4], cj2 = rs[6], cj3 = rs[8];
    float fj0 = rs[3], fj1 = rs[5], fj2 = rs[7];
    float trHc   = rs[10] + rs[12] + rs[14];
    float fi_lap = rs[11] + rs[13] + rs[15];
    // j_div = -D*tr(Hc) - K*(c_grd.Fi_grd + c*Fi_lap) + V*sum(c_grd); D=K=1,V=0.1
    float jdiv = -trHc - ((cj0 * fj0 + cj1 * fj1 + cj2 * fj2) + c * fi_lap)
                 + 0.1f * (cj0 + cj1 + cj2);
    out[gs]                 = c;        // c        (B,1)
    out[B + gs]             = cj3;      // c_t      (B,)   TDIM=3
    out[2 * B + gs * 3 + 0] = cj0;      // c_grd    (B,3)
    out[2 * B + gs * 3 + 1] = cj1;
    out[2 * B + gs * 3 + 2] = cj2;
    out[5 * B + gs]         = Fi;       // Fi       (B,1)
    out[6 * B + gs * 3 + 0] = fj0;      // Fi_grd   (B,3)
    out[6 * B + gs * 3 + 1] = fj1;
    out[6 * B + gs * 3 + 2] = fj2;
    out[9 * B + gs]         = fi_lap;   // Fi_lap   (B,1)
    out[10 * B + gs]        = jdiv;     // j_div    (B,1)
  }
}

extern "C" void kernel_launch(void* const* d_in, const int* in_sizes, int n_in,
                              void* d_out, int out_size, void* d_ws, size_t ws_size,
                              hipStream_t stream) {
  const float* x  = (const float*)d_in[0];
  const float* W0 = (const float*)d_in[1];
  const float* b0 = (const float*)d_in[2];
  const float* W1 = (const float*)d_in[3];
  const float* b1 = (const float*)d_in[4];
  const float* W2 = (const float*)d_in[5];
  const float* b2 = (const float*)d_in[6];
  const float* W3 = (const float*)d_in[7];
  const float* b3 = (const float*)d_in[8];
  const float* Wo = (const float*)d_in[9];
  const float* bo = (const float*)d_in[10];
  float* out = (float*)d_out;
  int B = in_sizes[0] / 4;

  bf16* Wq = (bf16*)d_ws;   // 3*65536 + 4096 bf16 = 392 KB packed weights

  int prep_n = 3 * NBF_L + NBF_O;
  prep_weights<<<(prep_n + 255) / 256, 256, 0, stream>>>(W1, W2, W3, Wo, Wq);
  mlp_hess<<<B / 16, 256, 0, stream>>>(x, W0, b0, Wq, b1, b2, b3, bo, out, B);
}
